// TrajectronPooling_34772055228499
// MI455X (gfx1250) — matrix operand order repaired and checked by reading, not compiled, not run on hardware
//
#include <hip/hip_runtime.h>

#define A_TOTAL 65536   // B*N
#define HDIM    256
#define DDIM    32
#define G4      1024    // 4*H
#define MT      4       // m-tiles (of 16 rows) per block -> 64 rows/block

typedef __attribute__((ext_vector_type(16))) __bf16 v16bf;
typedef __attribute__((ext_vector_type(8)))  __bf16 v8bf;
typedef __attribute__((ext_vector_type(8)))  float  v8f;

static __device__ inline v16bf bf_combine(v8bf lo, v8bf hi) {
  v16bf r;
#pragma unroll
  for (int e = 0; e < 8; ++e) { r[e] = lo[e]; r[e + 8] = hi[e]; }
  return r;
}

// B fragment (32x16 bf16) from N-major (transposed) weights wt[N][Krow].
static __device__ inline v16bf load_bfrag(const __bf16* __restrict__ wt, int Krow,
                                          int n, int K0, int hk) {
  const __bf16* p = wt + (size_t)n * Krow + K0 + hk * 8;
  v8bf lo = *(const v8bf*)(p);
  v8bf hi = *(const v8bf*)(p + 16);
  return bf_combine(lo, hi);
}

// A fragment (16x32 bf16) from an LDS tile laid out row-major [rows][Krow].
static __device__ inline v16bf lds_afrag(const __bf16* tile, int Krow,
                                         int row, int K0, int hk) {
  const __bf16* p = tile + row * Krow + K0 + hk * 8;
  return bf_combine(*(const v8bf*)(p), *(const v8bf*)(p + 16));
}

static __device__ inline float sigmoidf_(float x) { return 1.f / (1.f + expf(-x)); }

// ---------------- prep kernels ----------------

__global__ void init_total_kernel(float* total) {
  if (threadIdx.x < 4) total[threadIdx.x] = 0.f;
}

__global__ void stats_kernel(const float* __restrict__ obs1, const float* __restrict__ obs2,
                             float* __restrict__ sc, float* __restrict__ vis,
                             float* __restrict__ total) {
  __shared__ float lsum[4];
  if (threadIdx.x < 4) lsum[threadIdx.x] = 0.f;
  __syncthreads();
  int a = blockIdx.x * blockDim.x + threadIdx.x;
  float p0 = obs2[2 * a + 0], p1 = obs2[2 * a + 1];
  float q0 = obs1[2 * a + 0], q1 = obs1[2 * a + 1];
  float v0 = p0 - q0, v1 = p1 - q1;
  bool bad = (p0 != p0) || (p1 != p1) || (v0 != v0) || (v1 != v1);
  float s0 = bad ? 0.f : p0, s1 = bad ? 0.f : p1;
  float s2 = bad ? 0.f : v0, s3 = bad ? 0.f : v1;
  sc[4 * a + 0] = s0; sc[4 * a + 1] = s1; sc[4 * a + 2] = s2; sc[4 * a + 3] = s3;
  vis[a] = bad ? 0.f : 1.f;
  atomicAdd(&lsum[0], s0); atomicAdd(&lsum[1], s1);
  atomicAdd(&lsum[2], s2); atomicAdd(&lsum[3], s3);
  __syncthreads();
  if (threadIdx.x < 4) atomicAdd(&total[threadIdx.x], lsum[threadIdx.x]);
}

// dst[n*K + k] = (bf16) src[k*N + n]   (convert + transpose to N-major)
__global__ void transpose_bf16_kernel(const float* __restrict__ src, __bf16* __restrict__ dst,
                                      int K, int N) {
  int i = blockIdx.x * blockDim.x + threadIdx.x;
  if (i >= K * N) return;
  int n = i / K, k = i - n * K;
  dst[(size_t)n * K + k] = (__bf16)src[(size_t)k * N + n];
}

__global__ void f32_to_bf16_kernel(const float* __restrict__ src, __bf16* __restrict__ dst,
                                   int n) {
  int i = blockIdx.x * blockDim.x + threadIdx.x;
  if (i < n) dst[i] = (__bf16)src[i];
}

// feat = [sc, total - sc]; x = vis * relu(feat @ W_emb + b_emb)  -> bf16 [A,32]
__global__ void embed_kernel(const float* __restrict__ sc, const float* __restrict__ vis,
                             const float* __restrict__ total,
                             const float* __restrict__ W_emb, const float* __restrict__ b_emb,
                             __bf16* __restrict__ xb) {
  int a = blockIdx.x * blockDim.x + threadIdx.x;
  if (a >= A_TOTAL) return;
  float f[8];
  f[0] = sc[4 * a + 0]; f[1] = sc[4 * a + 1]; f[2] = sc[4 * a + 2]; f[3] = sc[4 * a + 3];
  f[4] = total[0] - f[0]; f[5] = total[1] - f[1]; f[6] = total[2] - f[2]; f[7] = total[3] - f[3];
  float v = vis[a];
#pragma unroll 4
  for (int j = 0; j < DDIM; ++j) {
    float acc = b_emb[j];
#pragma unroll
    for (int k = 0; k < 8; ++k) acc += f[k] * W_emb[k * DDIM + j];
    acc = acc > 0.f ? acc : 0.f;
    xb[(size_t)a * DDIM + j] = (__bf16)(acc * v);
  }
}

// ---------------- fused gates + LSTM + output GEMM ----------------
// 64 agent rows per block (MT=4 m-tiles), 8 waves. Wave w owns gate columns
// n = w*16 + 128*t (t = 0..7), so i/f/g/o of the same h (h, h+256, h+512, h+768)
// all live in that wave's accumulators -> LSTM nonlinearity stays in registers.
// h0/x tiles are staged in LDS once per block via gfx1250 async copies
// (global_load_async_to_lds_b128 + s_wait_asynccnt); the 32 KB h0 tile is
// reused as the h1 staging buffer for the output GEMM.
__global__ __launch_bounds__(256) void fused_lstm_kernel(
    const __bf16* __restrict__ xb,     // [A,32]
    const __bf16* __restrict__ h0b,    // [A,256]
    const __bf16* __restrict__ whht,   // [1024][256]  N-major
    const __bf16* __restrict__ wiht,   // [1024][32]   N-major
    const __bf16* __restrict__ woutt,  // [32][256]    N-major
    const float* __restrict__ b_ih, const float* __restrict__ b_hh,
    const float* __restrict__ c0, const float* __restrict__ b_out,
    float* __restrict__ out)           // [A,32]
{
  __shared__ __bf16 tileA[MT * 16 * HDIM];   // 32 KB: h0 tile, then reused for h1
  __shared__ __bf16 tileX[MT * 16 * DDIM];   // 4 KB : x tile
  const int tid  = threadIdx.x;
  const int wave = tid >> 5;
  const int lane = tid & 31;
  const int mlo  = lane & 15;
  const int hk   = lane >> 4;
  const int row0 = blockIdx.x * (MT * 16);

  // ---- async-stage h0 (32 KB) and x (4 KB) tiles into LDS ----
  {
    const char* gA = (const char*)(h0b + (size_t)row0 * HDIM);
    const char* gX = (const char*)(xb  + (size_t)row0 * DDIM);
    unsigned lA = (unsigned)(uintptr_t)&tileA[0];   // low 32 bits of flat shared
    unsigned lX = (unsigned)(uintptr_t)&tileX[0];   // addr == LDS byte offset
#pragma unroll
    for (int i = 0; i < 8; ++i) {
      unsigned o = (unsigned)tid * 16u + (unsigned)i * 4096u;
      asm volatile("global_load_async_to_lds_b128 %0, %1, off"
                   :: "v"(lA + o), "v"(gA + o) : "memory");
    }
    {
      unsigned o = (unsigned)tid * 16u;
      asm volatile("global_load_async_to_lds_b128 %0, %1, off"
                   :: "v"(lX + o), "v"(gX + o) : "memory");
    }
    asm volatile("s_wait_asynccnt 0x0" ::: "memory");
    __syncthreads();
  }

  v8f acc[MT][8];
#pragma unroll
  for (int mi = 0; mi < MT; ++mi)
#pragma unroll
    for (int t = 0; t < 8; ++t) { v8f z = {}; acc[mi][t] = z; }

  // gates += h0 @ W_hh   (K = 256, 8 k-steps of 32)
#pragma unroll
  for (int ks = 0; ks < 8; ++ks) {
    const int K0 = ks * 32;
    v16bf af[MT];
#pragma unroll
    for (int mi = 0; mi < MT; ++mi)
      af[mi] = lds_afrag(tileA, HDIM, mi * 16 + mlo, K0, hk);
    v16bf bfv[8];
#pragma unroll
    for (int t = 0; t < 8; ++t)
      bfv[t] = load_bfrag(whht, HDIM, wave * 16 + 128 * t + mlo, K0, hk);
#pragma unroll
    for (int t = 0; t < 8; ++t)
#pragma unroll
      for (int mi = 0; mi < MT; ++mi)
        acc[mi][t] = __builtin_amdgcn_wmma_f32_16x16x32_bf16(
            false, af[mi], false, bfv[t], (short)0, acc[mi][t], false, false);
  }

  // gates += x @ W_ih    (K = 32, single k-step)
  {
    v16bf af[MT];
#pragma unroll
    for (int mi = 0; mi < MT; ++mi)
      af[mi] = lds_afrag(tileX, DDIM, mi * 16 + mlo, 0, hk);
#pragma unroll
    for (int t = 0; t < 8; ++t) {
      v16bf bfrag = load_bfrag(wiht, DDIM, wave * 16 + 128 * t + mlo, 0, hk);
#pragma unroll
      for (int mi = 0; mi < MT; ++mi)
        acc[mi][t] = __builtin_amdgcn_wmma_f32_16x16x32_bf16(
            false, af[mi], false, bfrag, (short)0, acc[mi][t], false, false);
    }
  }

  __syncthreads();   // all waves done reading tileA/tileX; tileA becomes h1 buffer

  // bias + LSTM nonlinearity in registers (i/f/g/o pairs live in acc[mi][t0+2k])
#pragma unroll
  for (int t0 = 0; t0 < 2; ++t0) {
    const int h = wave * 16 + 128 * t0 + mlo;
    const float bi = b_ih[h]       + b_hh[h];
    const float bf = b_ih[h + 256] + b_hh[h + 256];
    const float bg = b_ih[h + 512] + b_hh[h + 512];
    const float bo = b_ih[h + 768] + b_hh[h + 768];
#pragma unroll
    for (int mi = 0; mi < MT; ++mi) {
#pragma unroll
      for (int r = 0; r < 8; ++r) {
        const int m = mi * 16 + r + hk * 8;
        float ig = acc[mi][t0 + 0][r] + bi;
        float fg = acc[mi][t0 + 2][r] + bf;
        float gg = acc[mi][t0 + 4][r] + bg;
        float og = acc[mi][t0 + 6][r] + bo;
        float c0v = c0[(size_t)(row0 + m) * HDIM + h];
        float c1 = sigmoidf_(fg) * c0v + sigmoidf_(ig) * tanhf(gg);
        float h1 = sigmoidf_(og) * tanhf(c1);
        tileA[m * HDIM + h] = (__bf16)h1;
      }
    }
  }
  __syncthreads();

  // out = h1 @ W_out + b_out   (64x32, K = 256); 8 tile jobs -> one per wave.
  {
    const int mi = wave >> 1;                 // m-tile 0..3
    const int n  = (wave & 1) * 16 + mlo;     // output column
    v8f oacc = {};
#pragma unroll
    for (int ks = 0; ks < 8; ++ks) {
      const int K0 = ks * 32;
      v16bf afrag = lds_afrag(tileA, HDIM, mi * 16 + mlo, K0, hk);
      v16bf bfrag = load_bfrag(woutt, HDIM, n, K0, hk);
      oacc = __builtin_amdgcn_wmma_f32_16x16x32_bf16(
          false, afrag, false, bfrag, (short)0, oacc, false, false);
    }
    const float bo = b_out[n];
#pragma unroll
    for (int r = 0; r < 8; ++r) {
      const int m = mi * 16 + r + hk * 8;
      out[(size_t)(row0 + m) * DDIM + n] = oacc[r] + bo;
    }
  }
}

// ---------------- launch ----------------

extern "C" void kernel_launch(void* const* d_in, const int* in_sizes, int n_in,
                              void* d_out, int out_size, void* d_ws, size_t ws_size,
                              hipStream_t stream) {
  (void)in_sizes; (void)n_in; (void)out_size; (void)ws_size;
  const float* obs1  = (const float*)d_in[0];
  const float* obs2  = (const float*)d_in[1];
  const float* h0    = (const float*)d_in[2];
  const float* c0    = (const float*)d_in[3];
  const float* W_emb = (const float*)d_in[4];
  const float* b_emb = (const float*)d_in[5];
  const float* W_ih  = (const float*)d_in[6];
  const float* W_hh  = (const float*)d_in[7];
  const float* b_ih  = (const float*)d_in[8];
  const float* b_hh  = (const float*)d_in[9];
  const float* W_out = (const float*)d_in[10];
  const float* b_out = (const float*)d_in[11];
  float* out = (float*)d_out;

  // workspace carve-up (256B aligned)
  char* ws = (char*)d_ws;
  size_t off = 0;
  auto take = [&](size_t bytes) { char* p = ws + off; off += (bytes + 255) & ~(size_t)255; return p; };
  float*  total = (float*) take(4 * sizeof(float));
  float*  sc    = (float*) take((size_t)A_TOTAL * 4 * sizeof(float));
  float*  vis   = (float*) take((size_t)A_TOTAL * sizeof(float));
  __bf16* xb    = (__bf16*)take((size_t)A_TOTAL * DDIM * sizeof(__bf16));
  __bf16* h0b   = (__bf16*)take((size_t)A_TOTAL * HDIM * sizeof(__bf16));
  __bf16* wiht  = (__bf16*)take((size_t)G4 * DDIM * sizeof(__bf16));
  __bf16* whht  = (__bf16*)take((size_t)G4 * HDIM * sizeof(__bf16));
  __bf16* woutt = (__bf16*)take((size_t)DDIM * HDIM * sizeof(__bf16));

  init_total_kernel<<<1, 64, 0, stream>>>(total);
  stats_kernel<<<A_TOTAL / 256, 256, 0, stream>>>(obs1, obs2, sc, vis, total);

  transpose_bf16_kernel<<<(DDIM * G4 + 255) / 256, 256, 0, stream>>>(W_ih,  wiht,  DDIM, G4);
  transpose_bf16_kernel<<<(HDIM * G4 + 255) / 256, 256, 0, stream>>>(W_hh,  whht,  HDIM, G4);
  transpose_bf16_kernel<<<(HDIM * DDIM + 255) / 256, 256, 0, stream>>>(W_out, woutt, HDIM, DDIM);

  f32_to_bf16_kernel<<<((size_t)A_TOTAL * HDIM + 255) / 256, 256, 0, stream>>>(h0, h0b, A_TOTAL * HDIM);

  embed_kernel<<<A_TOTAL / 256, 256, 0, stream>>>(sc, vis, total, W_emb, b_emb, xb);

  fused_lstm_kernel<<<A_TOTAL / (MT * 16), 256, 0, stream>>>(
      xb, h0b, whht, wiht, woutt, b_ih, b_hh, c0, b_out, out);
}
